// LongRangeModule_49237505082088
// MI455X (gfx1250) — compile-verified, matrix operand order checked by reading, not run on hardware
//
#include <hip/hip_runtime.h>

// ---------------------------------------------------------------------------
// Fused long-range aggregation for MI455X (gfx1250, wave32, WMMA).
//
//   cos   = |nrm(emb_i) @ nrm(emb_j)^T|        (8192x8192x256, bf16 WMMA)
//   valid = |i-j|>128 & mask_i & mask_j & cos>0.1 ; num_j = row counts
//   agg   = (valid*cos) @ x / num_j            (2x 8192x8192x512, bf16 WMMA)
//   out   = num_j>0 ? 0.5*(x+agg) : x
//
// Design (compute-bound, ~274 GFLOP after fusion):
//  * Prep: normalized bf16 panels nb_i/nb_j and bf16 transposed xT[b][d][l]
//    in d_ws (~25 MB) so the hot loop streams pure bf16 (halves L2 traffic,
//    no scale/convert VALU in the loop, transposed LDS writes eliminated).
//  * One block = 128-row i-tile x 128-col d-slice x BOTH batches; loops all j.
//    Cos tile computed once per block per j-tile (redundancy 8x -> 4x),
//    masked in registers, staged via LDS bf16, fed to the agg WMMA chain.
//  * Valid counts via ballot/popcount on the 16-lane C/D row halves.
//  * __builtin_prefetch (global_prefetch_b8) of next j-tile overlaps WMMA.
// ---------------------------------------------------------------------------

typedef __bf16 bf16_t;
typedef __attribute__((ext_vector_type(16))) __bf16 v16bf;
typedef __attribute__((ext_vector_type(8)))  __bf16 v8bf;
typedef __attribute__((ext_vector_type(8)))  float  v8f;

#define L_DIM 8192
#define D_DIM 512
#define E_DIM 256
#define B_DIM 2
#define CHUNK 128
#define CUTOFF 0.1f

#define BM 128            // i-tile rows
#define BN 128            // output column slice of D (per batch)
#define BJ 128            // j-tile (K of second GEMM)
#define LDE (E_DIM + 8)   // padded bf16 stride, K=256 panels (16B aligned)
#define LDJ (BJ + 8)      // padded bf16 stride, S / x tiles   (16B aligned)

__device__ __forceinline__ v8f wmma_bf16(v16bf a, v16bf b, v8f c) {
  return __builtin_amdgcn_wmma_f32_16x16x32_bf16(false, a, false, b,
                                                 (short)0, c, false, false);
}

// 16x32 bf16 A/B fragment from an LDS panel laid out [row][K] (row = matrix
// column for B). ISA 16-bit layout: lane<16 -> K {k0..+7, k0+16..+23},
// lane>=16 -> K {k0+8..+15, k0+24..+31}.  Two 16B ds_load_b128 per fragment.
__device__ __forceinline__ v16bf load_frag(const bf16_t* p, int ld,
                                           int row_base, int k0, int lane) {
  const int r = row_base + (lane & 15);
  const int k = k0 + ((lane >> 4) << 3);
  const bf16_t* q = p + r * ld + k;
  v8bf lo = *(const v8bf*)(q);
  v8bf hi = *(const v8bf*)(q + 16);
  v16bf f;
#pragma unroll
  for (int t = 0; t < 8; ++t) { f[t] = lo[t]; f[t + 8] = hi[t]; }
  return f;
}

// ---------------------------------------------------------------------------
// Prep 1: per-row L2-normalize an (L x E) table into bf16.
// ---------------------------------------------------------------------------
__global__ __launch_bounds__(256)
void lr_norm_kernel(const float* __restrict__ emb, bf16_t* __restrict__ nb) {
  __shared__ float red[8];
  __shared__ float s_inv;
  const int row = blockIdx.x;
  const int tid = threadIdx.x;           // 256 == E_DIM
  const float v = emb[(size_t)row * E_DIM + tid];
  float s = v * v;
#pragma unroll
  for (int off = 16; off > 0; off >>= 1) s += __shfl_xor(s, off, 32);
  if ((tid & 31) == 0) red[tid >> 5] = s;
  __syncthreads();
  if (tid == 0) {
    float t = 0.f;
#pragma unroll
    for (int i = 0; i < 8; ++i) t += red[i];
    s_inv = 1.0f / fmaxf(sqrtf(t), 1e-8f);
  }
  __syncthreads();
  nb[(size_t)row * E_DIM + tid] = (bf16_t)(v * s_inv);
}

// ---------------------------------------------------------------------------
// Prep 2: xT[b][d][l] = (bf16) x[b][l][d]   (LDS-tiled, coalesced both ways)
// ---------------------------------------------------------------------------
__global__ __launch_bounds__(256)
void lr_xt_kernel(const float* __restrict__ x, bf16_t* __restrict__ xT) {
  __shared__ float t[32][33];
  const int l0 = blockIdx.x * 32, d0 = blockIdx.y * 32, b = blockIdx.z;
  const int tx = threadIdx.x, ty = threadIdx.y;   // 32 x 8
#pragma unroll
  for (int k = 0; k < 4; ++k)
    t[ty + 8 * k][tx] = x[((size_t)b * L_DIM + l0 + ty + 8 * k) * D_DIM + d0 + tx];
  __syncthreads();
#pragma unroll
  for (int k = 0; k < 4; ++k)
    xT[((size_t)b * D_DIM + d0 + ty + 8 * k) * L_DIM + l0 + tx] =
        (bf16_t)t[tx][ty + 8 * k];
}

// ---------------------------------------------------------------------------
// Fused kernel. Grid: (L/BM, D/BN). Block: 256 thr = 8 wave32 (4x2 waves).
// ---------------------------------------------------------------------------
__global__ __launch_bounds__(256)
void lr_fused_kernel(const float* __restrict__ x,
                     const int* __restrict__ mask,
                     const bf16_t* __restrict__ nb_i,
                     const bf16_t* __restrict__ nb_j,
                     const bf16_t* __restrict__ xT,
                     float* __restrict__ out) {
  __shared__ __align__(16) bf16_t s_ni[BM * LDE];          // resident i-panel
  __shared__ __align__(16) bf16_t s_nj[BJ * LDE];          // j-panel
  __shared__ __align__(16) bf16_t s_x [B_DIM][BN * LDJ];   // x tiles [n][j]
  __shared__ __align__(16) bf16_t s_s [BM * LDJ];          // masked cos tile
  __shared__ int s_mi[BM];
  __shared__ int s_mj[BJ];
  __shared__ int s_cnt[BM];

  const int i0    = blockIdx.x * BM;
  const int n0    = blockIdx.y * BN;
  const int tid   = threadIdx.x;
  const int lane  = tid & 31;
  const int wid   = tid >> 5;
  const int wrow  = (wid & 3) * 32;      // wave's i offset
  const int wcol  = (wid >> 2) * 64;     // wave's n (or j) offset
  const int halfo = (lane >> 4) << 3;    // +8 row offset for upper lane half
  const int nlan  = lane & 15;
  const unsigned long long halfmask = 0xffffULL << (lane & 16);

  // resident normalized emb_i panel (pure bf16 copy, 16B granules)
  for (int idx = tid; idx < BM * (E_DIM / 8); idx += 256) {
    const int row = idx >> 5;                      // 32 granules per row
    const int c   = (idx & 31) << 3;
    *(v8bf*)(s_ni + row * LDE + c) =
        *(const v8bf*)(nb_i + (size_t)(i0 + row) * E_DIM + c);
  }
  if (tid < BM) s_mi[tid] = mask[i0 + tid];

  v8f acc2[B_DIM][2][4];                 // persistent agg accumulators
#pragma unroll
  for (int bb = 0; bb < B_DIM; ++bb)
#pragma unroll
    for (int mi = 0; mi < 2; ++mi)
#pragma unroll
      for (int nt = 0; nt < 4; ++nt)
        acc2[bb][mi][nt] = v8f{0.f, 0.f, 0.f, 0.f, 0.f, 0.f, 0.f, 0.f};

  int cnt[2][8];                         // per-half-row valid counts
#pragma unroll
  for (int mi = 0; mi < 2; ++mi)
#pragma unroll
    for (int r = 0; r < 8; ++r) cnt[mi][r] = 0;

  // ================= main loop over all j =================
  for (int j0 = 0; j0 < L_DIM; j0 += BJ) {
    __syncthreads();                     // prev GEMMs done before restaging

    // stage normalized emb_j panel (bf16 copy)
    for (int idx = tid; idx < BJ * (E_DIM / 8); idx += 256) {
      const int row = idx >> 5;
      const int c   = (idx & 31) << 3;
      *(v8bf*)(s_nj + row * LDE + c) =
          *(const v8bf*)(nb_j + (size_t)(j0 + row) * E_DIM + c);
    }
    // stage x tiles for both batches: xT rows are contiguous in j
    for (int idx = tid; idx < B_DIM * BN * (BJ / 8); idx += 256) {
      const int bb  = idx >> 11;                   // BN*BJ/8 = 2048 per batch
      const int row = (idx >> 4) & (BN - 1);       // n local
      const int c   = (idx & 15) << 3;             // j local granule
      *(v8bf*)(&s_x[bb][row * LDJ + c]) =
          *(const v8bf*)(xT + ((size_t)bb * D_DIM + n0 + row) * L_DIM + j0 + c);
    }
    if (tid < BJ) s_mj[tid] = mask[j0 + tid];
    __syncthreads();

    // prefetch next j-tile while this tile's 128 WMMAs run
    if (j0 + BJ < L_DIM) {
      const char* pnj = (const char*)(nb_j + (size_t)(j0 + BJ) * E_DIM);
      __builtin_prefetch(pnj + tid * 128, 0, 3);           // 32 KB
      __builtin_prefetch(pnj + 32768 + tid * 128, 0, 3);   // 32 KB
      const int pb = tid >> 7, pr = tid & 127;             // one xT row each
      const char* px = (const char*)(xT + ((size_t)pb * D_DIM + n0 + pr) * L_DIM
                                     + (j0 + BJ));
      __builtin_prefetch(px, 0, 3);
      __builtin_prefetch(px + 128, 0, 3);
    }

    // ---- GEMM1: cos tile (128 x 128, K = 256) ----
    v8f acc1[2][4];
#pragma unroll
    for (int mi = 0; mi < 2; ++mi)
#pragma unroll
      for (int nt = 0; nt < 4; ++nt)
        acc1[mi][nt] = v8f{0.f, 0.f, 0.f, 0.f, 0.f, 0.f, 0.f, 0.f};

#pragma unroll
    for (int ks = 0; ks < E_DIM; ks += 32) {
      v16bf a0 = load_frag(s_ni, LDE, wrow +  0, ks, lane);
      v16bf a1 = load_frag(s_ni, LDE, wrow + 16, ks, lane);
      v16bf b0 = load_frag(s_nj, LDE, wcol +  0, ks, lane);
      v16bf b1 = load_frag(s_nj, LDE, wcol + 16, ks, lane);
      v16bf b2 = load_frag(s_nj, LDE, wcol + 32, ks, lane);
      v16bf b3 = load_frag(s_nj, LDE, wcol + 48, ks, lane);
      acc1[0][0] = wmma_bf16(a0, b0, acc1[0][0]);
      acc1[0][1] = wmma_bf16(a0, b1, acc1[0][1]);
      acc1[0][2] = wmma_bf16(a0, b2, acc1[0][2]);
      acc1[0][3] = wmma_bf16(a0, b3, acc1[0][3]);
      acc1[1][0] = wmma_bf16(a1, b0, acc1[1][0]);
      acc1[1][1] = wmma_bf16(a1, b1, acc1[1][1]);
      acc1[1][2] = wmma_bf16(a1, b2, acc1[1][2]);
      acc1[1][3] = wmma_bf16(a1, b3, acc1[1][3]);
    }

    // ---- mask + ballot-count + write bf16 S tile ----
#pragma unroll
    for (int mi = 0; mi < 2; ++mi) {
#pragma unroll
      for (int nt = 0; nt < 4; ++nt) {
#pragma unroll
        for (int r = 0; r < 8; ++r) {
          const int ml = wrow + mi * 16 + r + halfo;
          const int nl = wcol + nt * 16 + nlan;
          const float c = fabsf(acc1[mi][nt][r]);
          const int gi = i0 + ml, gj = j0 + nl;
          int dist = gi - gj; dist = dist < 0 ? -dist : dist;
          const bool valid = (dist > CHUNK) && (s_mi[ml] != 0) &&
                             (s_mj[nl] != 0) && (c > CUTOFF);
          s_s[ml * LDJ + nl] = (bf16_t)(valid ? c : 0.f);
          cnt[mi][r] += __popcll(__ballot(valid) & halfmask);
        }
      }
    }
    __syncthreads();

    // ---- GEMM2: acc2[b] += S_tile @ x_tile[b]  (K = 128 over j) ----
#pragma unroll
    for (int ks = 0; ks < BJ; ks += 32) {
      v16bf a0 = load_frag(s_s, LDJ, wrow +  0, ks, lane);
      v16bf a1 = load_frag(s_s, LDJ, wrow + 16, ks, lane);
#pragma unroll
      for (int bb = 0; bb < B_DIM; ++bb) {
        v16bf b0 = load_frag(s_x[bb], LDJ, wcol +  0, ks, lane);
        v16bf b1 = load_frag(s_x[bb], LDJ, wcol + 16, ks, lane);
        v16bf b2 = load_frag(s_x[bb], LDJ, wcol + 32, ks, lane);
        v16bf b3 = load_frag(s_x[bb], LDJ, wcol + 48, ks, lane);
        acc2[bb][0][0] = wmma_bf16(a0, b0, acc2[bb][0][0]);
        acc2[bb][0][1] = wmma_bf16(a0, b1, acc2[bb][0][1]);
        acc2[bb][0][2] = wmma_bf16(a0, b2, acc2[bb][0][2]);
        acc2[bb][0][3] = wmma_bf16(a0, b3, acc2[bb][0][3]);
        acc2[bb][1][0] = wmma_bf16(a1, b0, acc2[bb][1][0]);
        acc2[bb][1][1] = wmma_bf16(a1, b1, acc2[bb][1][1]);
        acc2[bb][1][2] = wmma_bf16(a1, b2, acc2[bb][1][2]);
        acc2[bb][1][3] = wmma_bf16(a1, b3, acc2[bb][1][3]);
      }
    }
  }

  // ---- gather per-row counts (all 16 lanes of a half hold the same sum) ----
  __syncthreads();
  if (tid < BM) s_cnt[tid] = 0;
  __syncthreads();
  if (nlan == 0) {
#pragma unroll
    for (int mi = 0; mi < 2; ++mi)
#pragma unroll
      for (int r = 0; r < 8; ++r)
        atomicAdd(&s_cnt[wrow + mi * 16 + r + halfo], cnt[mi][r]);
  }
  __syncthreads();

  // ---- epilogue: out = cnt>0 ? 0.5*(x + agg/cnt) : x ----
#pragma unroll
  for (int bb = 0; bb < B_DIM; ++bb) {
#pragma unroll
    for (int mi = 0; mi < 2; ++mi) {
#pragma unroll
      for (int nt = 0; nt < 4; ++nt) {
#pragma unroll
        for (int r = 0; r < 8; ++r) {
          const int ml = wrow + mi * 16 + r + halfo;
          const int nl = wcol + nt * 16 + nlan;
          const size_t g = ((size_t)bb * L_DIM + (i0 + ml)) * D_DIM + (n0 + nl);
          const float xv = x[g];
          const int   c  = s_cnt[ml];
          out[g] = (c > 0) ? 0.5f * (xv + acc2[bb][mi][nt][r] / (float)c) : xv;
        }
      }
    }
  }
}

// ---------------------------------------------------------------------------
extern "C" void kernel_launch(void* const* d_in, const int* in_sizes, int n_in,
                              void* d_out, int out_size, void* d_ws, size_t ws_size,
                              hipStream_t stream) {
  (void)in_sizes; (void)n_in; (void)out_size; (void)ws_size;
  const float* x     = (const float*)d_in[0];
  const int*   mask  = (const int*)d_in[1];
  const float* emb_i = (const float*)d_in[2];
  const float* emb_j = (const float*)d_in[3];
  float* out = (float*)d_out;

  // workspace layout (bf16 staging buffers, ~25 MB total)
  char* ws = (char*)d_ws;
  bf16_t* nb_i = (bf16_t*)(ws);                                   // 4 MB
  bf16_t* nb_j = (bf16_t*)(ws + (size_t)L_DIM * E_DIM * 2);       // 4 MB
  bf16_t* xT   = (bf16_t*)(ws + (size_t)L_DIM * E_DIM * 4);       // 16.8 MB

  lr_norm_kernel<<<L_DIM, 256, 0, stream>>>(emb_i, nb_i);
  lr_norm_kernel<<<L_DIM, 256, 0, stream>>>(emb_j, nb_j);

  dim3 tgrid(L_DIM / 32, D_DIM / 32, B_DIM);
  lr_xt_kernel<<<tgrid, dim3(32, 8), 0, stream>>>(x, xT);

  dim3 grid(L_DIM / BM, D_DIM / BN);          // 64 x 4 = 256 blocks
  lr_fused_kernel<<<grid, 256, 0, stream>>>(x, mask, nb_i, nb_j, xT, out);
}
